// DecoderAttention_74491912782418
// MI455X (gfx1250) — matrix operand order repaired and checked by reading, not compiled, not run on hardware
//
#include <hip/hip_runtime.h>
#include <hip/hip_bf16.h>

// ---------------------------------------------------------------------------
// SketchRNN-style LSTM decoder on MI455X (gfx1250, wave32, WMMA bf16).
// Per step: gates[256x8192] = A[256x2208] * Wc^T, fused LSTM cell epilogue.
// Wc (36MB bf16) is L2-resident (192MB L2); block tile 128x32(x4 gates) with
// LDS-staged, double-buffered A/B chunks -> Wc read only 2x/step from L2.
// ---------------------------------------------------------------------------

typedef __bf16 bf16_t;
typedef __attribute__((ext_vector_type(16))) __bf16 v16bf;
typedef __attribute__((ext_vector_type(8)))  __bf16 v8bf;
typedef __attribute__((ext_vector_type(8)))  float  v8f;

#define BATCH 256
#define HID   2048
#define NZ    128
#define NX    5
#define NIN   133            // NZ + NX
#define G4    8192           // 4*HID
#define KA    2208           // padded K: HID + NX + NZ = 2181 -> 2208 (mult of 32)
#define KX0   2048           // x columns start inside A / Wc
#define KZ0   2053           // z columns start
#define KREAL 2181
#define NMIX  20
#define NOUT  123            // 6*M + 3
#define NOUTP 128            // padded output dim
#define BM    128            // batch rows per block
#define BJ    32             // hidden units per block (x4 gates)
#define LROW  40             // LDS row stride in elements (80B, 16B-aligned, bank-skewed)
#define NCHUNK (KA / 32)     // 69

__device__ __forceinline__ float sigf(float x) { return 1.0f / (1.0f + __expf(-x)); }
// branch-free tanh: 2*sigmoid(2x)-1 (saturates correctly at +-1)
__device__ __forceinline__ float tanh_fast(float x) {
    return __builtin_fmaf(2.0f, 1.0f / (1.0f + __expf(-2.0f * x)), -1.0f);
}

// A fragment (16x32 bf16) from row-major global memory, leading dim lda.
// Lanes 0-15: row m0+l, K=k0+0..15; lanes 16-31: K=k0+16..31.
__device__ __forceinline__ v16bf load_a_frag_g(const bf16_t* A, int lda, int m0, int k0, int lane) {
    int l = lane & 15, hi = lane >> 4;
    const bf16_t* p = A + (size_t)(m0 + l) * lda + k0 + hi * 16;
    v8bf lo = *(const v8bf*)p;
    v8bf hh = *(const v8bf*)(p + 8);
    v16bf a;
#pragma unroll
    for (int i = 0; i < 8; ++i) { a[i] = lo[i]; a[8 + i] = hh[i]; }
    return a;
}

// B fragment (32x16 bf16): B[k][n] = Wm[n0+n][k0+k], Wm row-major (ldw).
// Lanes 0-15: col n, K=k0+0..15; lanes 16-31: K=k0+16..31.
__device__ __forceinline__ v16bf load_b_frag_g(const bf16_t* Wm, int ldw, int n0, int k0, int lane) {
    int n = lane & 15, hi = lane >> 4;
    const bf16_t* p = Wm + (size_t)(n0 + n) * ldw + k0 + hi * 16;
    v8bf lo = *(const v8bf*)p;
    v8bf hh = *(const v8bf*)(p + 8);
    v16bf b;
#pragma unroll
    for (int i = 0; i < 8; ++i) { b[i] = lo[i]; b[8 + i] = hh[i]; }
    return b;
}

// ---------------------------------------------------------------------------
// Prep kernels (run every launch; deterministic)
// ---------------------------------------------------------------------------
__global__ __launch_bounds__(256) void pack_wc_kernel(
    const float* __restrict__ W_ih, const float* __restrict__ W_hh, bf16_t* __restrict__ Wc) {
    size_t idx = (size_t)blockIdx.x * 256 + threadIdx.x;
    if (idx >= (size_t)G4 * KA) return;
    int k = (int)(idx % KA);
    int g = (int)(idx / KA);
    float v;
    if (k < KX0)        v = W_hh[(size_t)g * HID + k];
    else if (k < KZ0)   v = W_ih[(size_t)g * NIN + (k - KX0)];        // x cols 0..4
    else if (k < KREAL) v = W_ih[(size_t)g * NIN + NX + (k - KZ0)];   // z cols 5..132
    else                v = 0.0f;
    Wc[idx] = (bf16_t)v;
}

__global__ __launch_bounds__(256) void pack_wyp_kernel(
    const float* __restrict__ Wy, bf16_t* __restrict__ Wyp) {
    size_t idx = (size_t)blockIdx.x * 256 + threadIdx.x;
    if (idx >= (size_t)NOUTP * HID) return;
    int k = (int)(idx % HID);
    int g = (int)(idx / HID);
    Wyp[idx] = (bf16_t)((g < NOUT) ? Wy[(size_t)g * HID + k] : 0.0f);
}

__global__ __launch_bounds__(256) void bias_kernel(
    const float* __restrict__ b_ih, const float* __restrict__ b_hh, float* __restrict__ bsum) {
    int i = blockIdx.x * 256 + threadIdx.x;
    if (i < G4) bsum[i] = b_ih[i] + b_hh[i];
}

__global__ __launch_bounds__(256) void init_a_kernel(
    const float* __restrict__ x, const float* __restrict__ z, const float* __restrict__ h0,
    bf16_t* __restrict__ A0, bf16_t* __restrict__ A1) {
    int idx = blockIdx.x * 256 + threadIdx.x;
    if (idx >= BATCH * KA) return;
    int k = idx % KA, b = idx / KA;
    float v0, v1;
    if (k < KX0)        { v0 = h0[(size_t)b * HID + k];           v1 = 0.0f; }
    else if (k < KZ0)   { v0 = x[(size_t)b * NX + (k - KX0)];     v1 = 0.0f; }   // x[t=0]
    else if (k < KREAL) { v0 = v1 = z[(size_t)b * NZ + (k - KZ0)]; }
    else                { v0 = v1 = 0.0f; }
    A0[idx] = (bf16_t)v0;
    A1[idx] = (bf16_t)v1;
}

__global__ __launch_bounds__(256) void init_state_kernel(
    const float* __restrict__ h0, const float* __restrict__ c0,
    float* __restrict__ hout, float* __restrict__ cstate) {
    int i = blockIdx.x * 256 + threadIdx.x;
    if (i < BATCH * HID) { hout[i] = h0[i]; cstate[i] = c0[i]; }
}

// ---------------------------------------------------------------------------
// One LSTM step. Grid (HID/BJ=64, BATCH/BM=2), block 256 = 8 waves.
// Wave (mt = w>>1 in 0..3, jt = w&1): two 16-row M tiles (rows mt*32..+31) x
// one 16-col J tile, computed for all FOUR gates -> 8 WMMAs per 32-K chunk,
// LSTM cell fuses into the epilogue.
// A and B chunks are staged in LDS (double-buffered, register-prefetched):
// each thread stages 4x16B pieces; rows 0..127 = A rows, 128..255 = B rows.
// ---------------------------------------------------------------------------
__global__ __launch_bounds__(256) void lstm_step_kernel(
    const bf16_t* __restrict__ Acur, bf16_t* __restrict__ Anext,
    const bf16_t* __restrict__ Wc, const float* __restrict__ bsum,
    float* __restrict__ cstate, float* __restrict__ hout) {
    __shared__ bf16_t lds[2][256 * LROW];   // 2 x 20KB

    int tid = threadIdx.x;
    int lane = tid & 31, wave = tid >> 5;
    int mbase = blockIdx.y * BM;
    int jb0 = blockIdx.x * BJ;
    int mt = wave >> 1;                 // 0..3 (32-row group)
    int jt = wave & 1;                  // 0..1 (16-col tile)
    int j0 = jb0 + jt * 16;
    int m0 = mbase + mt * 32;

    // ---- staging map: 4 pieces (16B each) per thread ----
    const bf16_t* gsrc[4];
    int loff[4];
#pragma unroll
    for (int i = 0; i < 4; ++i) {
        int q = tid * 4 + i;            // 0..1023
        int row = q >> 2;               // 0..255
        int part = q & 3;               // 16B piece within 64B row chunk
        if (row < BM) {                 // A rows
            gsrc[i] = Acur + (size_t)(mbase + row) * KA + part * 8;
        } else {                        // B rows: gate*32 + col
            int r = row - BM;
            gsrc[i] = Wc + (size_t)((r >> 5) * HID + jb0 + (r & 31)) * KA + part * 8;
        }
        loff[i] = row * LROW + part * 8;
    }

    v8bf st[4];
#pragma unroll
    for (int i = 0; i < 4; ++i) st[i] = *(const v8bf*)gsrc[i];   // prefetch chunk 0

    v8f acc[2][4];
#pragma unroll
    for (int s = 0; s < 2; ++s)
#pragma unroll
        for (int g = 0; g < 4; ++g) acc[s][g] = (v8f){};

    int l = lane & 15, hi = lane >> 4;

#pragma unroll 1
    for (int c = 0; c < NCHUNK; ++c) {
        int buf = c & 1;
        bf16_t* lb = lds[buf];
#pragma unroll
        for (int i = 0; i < 4; ++i) *(v8bf*)&lb[loff[i]] = st[i];
        __syncthreads();
        if (c + 1 < NCHUNK) {
#pragma unroll
            for (int i = 0; i < 4; ++i) st[i] = *(const v8bf*)(gsrc[i] + (size_t)(c + 1) * 32);
        }
        // B fragments for the 4 gates (reused across both M sub-tiles)
        v16bf bfr[4];
#pragma unroll
        for (int g = 0; g < 4; ++g) {
            int row = BM + g * 32 + jt * 16 + (lane & 15);
            bfr[g] = *(const v16bf*)&lb[row * LROW + hi * 16];   // 32B LDS read
        }
#pragma unroll
        for (int s = 0; s < 2; ++s) {
            int arow = mt * 32 + s * 16 + l;
            const bf16_t* ap = &lb[arow * LROW + hi * 16];
            v8bf alo = *(const v8bf*)ap;
            v8bf ahh = *(const v8bf*)(ap + 8);
            v16bf a;
#pragma unroll
            for (int i = 0; i < 8; ++i) { a[i] = alo[i]; a[8 + i] = ahh[i]; }
            acc[s][0] = __builtin_amdgcn_wmma_f32_16x16x32_bf16(false, a, false, bfr[0], (short)0, acc[s][0], false, false);
            acc[s][1] = __builtin_amdgcn_wmma_f32_16x16x32_bf16(false, a, false, bfr[1], (short)0, acc[s][1], false, false);
            acc[s][2] = __builtin_amdgcn_wmma_f32_16x16x32_bf16(false, a, false, bfr[2], (short)0, acc[s][2], false, false);
            acc[s][3] = __builtin_amdgcn_wmma_f32_16x16x32_bf16(false, a, false, bfr[3], (short)0, acc[s][3], false, false);
        }
        // double-buffered: next iteration writes the other LDS buffer, and the
        // barrier above (next iter) orders reuse of this one.
    }

    // ---- fused LSTM cell epilogue ----
    // C/D layout: elem e, lane L -> row +e+(L>=16)*8, col (L&15)
    int j = j0 + l;
    float bi  = bsum[0 * HID + j];
    float bf_ = bsum[1 * HID + j];
    float bg  = bsum[2 * HID + j];
    float bo  = bsum[3 * HID + j];
#pragma unroll
    for (int s = 0; s < 2; ++s) {
#pragma unroll
        for (int e = 0; e < 8; ++e) {
            int m = m0 + s * 16 + e + hi * 8;
            size_t idx = (size_t)m * HID + j;
            float iv = sigf(acc[s][0][e] + bi);
            float fv = sigf(acc[s][1][e] + bf_);
            float gv = tanh_fast(acc[s][2][e] + bg);
            float ov = sigf(acc[s][3][e] + bo);
            float cn = fv * cstate[idx] + iv * gv;
            float hn = ov * tanh_fast(cn);
            cstate[idx] = cn;
            hout[idx] = hn;
            Anext[(size_t)m * KA + j] = (bf16_t)hn;   // h cols of next ping-pong buffer
        }
    }
}

// ---------------------------------------------------------------------------
// Output head for step t: y = h_t * Wy^T + by, mixture head nonlinearities.
// Grid: 16 blocks (batch/16); block 256 = 8 waves; wave w -> N tile w*16.
// Also stages x[t+1] into the A buffer for the next step.
// ---------------------------------------------------------------------------
__global__ __launch_bounds__(256) void head_kernel(
    bf16_t* __restrict__ Acur, const bf16_t* __restrict__ Wyp,
    const float* __restrict__ by, const float* __restrict__ x,
    float* __restrict__ out, int t, int T) {
    __shared__ float ylds[16][NOUTP + 1];
    int lane = threadIdx.x & 31;
    int wave = threadIdx.x >> 5;
    int m0 = blockIdx.x * 16;
    int n0 = wave * 16;

    v8f acc = {};
#pragma unroll 1
    for (int k0 = 0; k0 < HID; k0 += 32) {
        v16bf a = load_a_frag_g(Acur, KA, m0, k0, lane);     // h occupies cols [0,2048)
        v16bf b = load_b_frag_g(Wyp, HID, n0, k0, lane);
        acc = __builtin_amdgcn_wmma_f32_16x16x32_bf16(false, a, false, b, (short)0, acc, false, false);
    }
    int l = lane & 15, hi = lane >> 4;
    int col = n0 + l;
    float byv = (col < NOUT) ? by[col] : 0.0f;
#pragma unroll
    for (int e = 0; e < 8; ++e) ylds[e + hi * 8][col] = acc[e] + byv;
    __syncthreads();

    if (threadIdx.x < 16) {
        int r = threadIdx.x;
        int b_ = m0 + r;
        size_t TB = (size_t)T * BATCH;
        size_t base = (size_t)t * BATCH + b_;
        float* p_pi  = out;
        float* p_mux = out + TB * NMIX;
        float* p_muy = out + TB * NMIX * 2;
        float* p_sx  = out + TB * NMIX * 3;
        float* p_sy  = out + TB * NMIX * 4;
        float* p_rho = out + TB * NMIX * 5;
        float* p_q   = out + TB * NMIX * 6;

        // pi = softmax(y[0:20])
        float mx = -3.4e38f;
#pragma unroll
        for (int k = 0; k < NMIX; ++k) mx = fmaxf(mx, ylds[r][k]);
        float ex[NMIX], s = 0.0f;
#pragma unroll
        for (int k = 0; k < NMIX; ++k) { ex[k] = __expf(ylds[r][k] - mx); s += ex[k]; }
        float inv = 1.0f / s;
#pragma unroll
        for (int k = 0; k < NMIX; ++k) p_pi[base * NMIX + k] = ex[k] * inv;
#pragma unroll
        for (int k = 0; k < NMIX; ++k) p_mux[base * NMIX + k] = ylds[r][NMIX + k];
#pragma unroll
        for (int k = 0; k < NMIX; ++k) p_muy[base * NMIX + k] = ylds[r][2 * NMIX + k];
#pragma unroll
        for (int k = 0; k < NMIX; ++k) p_sx[base * NMIX + k] = __expf(ylds[r][3 * NMIX + k]);  // sqrt(TAU)=1
#pragma unroll
        for (int k = 0; k < NMIX; ++k) p_sy[base * NMIX + k] = __expf(ylds[r][4 * NMIX + k]);
#pragma unroll
        for (int k = 0; k < NMIX; ++k) {
            float rv = tanh_fast(ylds[r][5 * NMIX + k]);
            p_rho[base * NMIX + k] = fminf(fmaxf(rv, -1.0f + 1e-6f), 1.0f - 1e-6f);
        }
        // q = softmax(y[120:123])
        float q0 = ylds[r][120], q1 = ylds[r][121], q2 = ylds[r][122];
        float qm = fmaxf(q0, fmaxf(q1, q2));
        float e0 = __expf(q0 - qm), e1 = __expf(q1 - qm), e2 = __expf(q2 - qm);
        float qi = 1.0f / (e0 + e1 + e2);
        p_q[base * 3 + 0] = e0 * qi;
        p_q[base * 3 + 1] = e1 * qi;
        p_q[base * 3 + 2] = e2 * qi;

        // Stage x[t+1] into the buffer step t+1 will read (this one).
        if (t + 1 < T) {
            const float* xp = x + ((size_t)(t + 1) * BATCH + b_) * NX;
            bf16_t* ax = Acur + (size_t)b_ * KA + KX0;
#pragma unroll
            for (int k = 0; k < NX; ++k) ax[k] = (bf16_t)xp[k];
        }
    }
}

// ---------------------------------------------------------------------------
// Host side
// ---------------------------------------------------------------------------
extern "C" void kernel_launch(void* const* d_in, const int* in_sizes, int n_in,
                              void* d_out, int out_size, void* d_ws, size_t ws_size,
                              hipStream_t stream) {
    const float* x    = (const float*)d_in[0];
    const float* z    = (const float*)d_in[1];
    const float* h0   = (const float*)d_in[2];
    const float* c0   = (const float*)d_in[3];
    const float* W_ih = (const float*)d_in[4];
    const float* W_hh = (const float*)d_in[5];
    const float* b_ih = (const float*)d_in[6];
    const float* b_hh = (const float*)d_in[7];
    const float* Wy   = (const float*)d_in[8];
    const float* by   = (const float*)d_in[9];
    float* out = (float*)d_out;

    const int T = in_sizes[0] / (BATCH * NX);   // 200

    // Workspace layout (256B-aligned offsets), ~39 MB total
    char* ws = (char*)d_ws;
    const size_t WS_WC  = 0;                                   // 8192*2208*2 = 36,175,872
    const size_t WS_WYP = WS_WC  + (size_t)G4 * KA * 2;        // 128*2048*2  = 524,288
    const size_t WS_A0  = WS_WYP + (size_t)NOUTP * HID * 2;    // 256*2208*2  = 1,130,496
    const size_t WS_A1  = WS_A0  + (size_t)BATCH * KA * 2;
    const size_t WS_BS  = WS_A1  + (size_t)BATCH * KA * 2;     // 8192*4
    bf16_t* Wc  = (bf16_t*)(ws + WS_WC);
    bf16_t* Wyp = (bf16_t*)(ws + WS_WYP);
    bf16_t* A0  = (bf16_t*)(ws + WS_A0);
    bf16_t* A1  = (bf16_t*)(ws + WS_A1);
    float*  bs  = (float*)(ws + WS_BS);

    // Final h/c live directly in their d_out slots (c doubles as running state)
    size_t TB = (size_t)T * BATCH;
    float* hout   = out + TB * NOUT;              // after pi..q: T*B*123
    float* cstate = hout + (size_t)BATCH * HID;

    { size_t n = (size_t)G4 * KA;      pack_wc_kernel <<<(unsigned)((n + 255) / 256), 256, 0, stream>>>(W_ih, W_hh, Wc); }
    { size_t n = (size_t)NOUTP * HID;  pack_wyp_kernel<<<(unsigned)((n + 255) / 256), 256, 0, stream>>>(Wy, Wyp); }
    bias_kernel<<<(G4 + 255) / 256, 256, 0, stream>>>(b_ih, b_hh, bs);
    { int n = BATCH * KA;              init_a_kernel  <<<(n + 255) / 256, 256, 0, stream>>>(x, z, h0, A0, A1); }
    init_state_kernel<<<(BATCH * HID + 255) / 256, 256, 0, stream>>>(h0, c0, hout, cstate);

    for (int t = 0; t < T; ++t) {
        bf16_t* Ain  = (t & 1) ? A1 : A0;
        bf16_t* Anew = (t & 1) ? A0 : A1;
        lstm_step_kernel<<<dim3(HID / BJ, BATCH / BM), 256, 0, stream>>>(Ain, Anew, Wc, bs, cstate, hout);
        head_kernel<<<BATCH / 16, 256, 0, stream>>>(Anew, Wyp, by, x, out, t, T);
    }
}
